// scFuMamba_stage1_80865644249259
// MI455X (gfx1250) — matrix operand order repaired and checked by reading, not compiled, not run on hardware
//
#include <hip/hip_runtime.h>
#include <math.h>

// ---------------------------------------------------------------------------
// scFuMamba (Mamba MAE, RNA+ADT) for gfx1250.
// Dense projections: v_wmma_f32_16x16x32_f16 with a pre-packed, zero-padded,
// transposed f16 B (no predicated loads anywhere in the GEMM inner loop).
// Per K-step: all 8 B b128 loads issue first (sched barrier keeps them
// grouped), then 4 back-to-back WMMAs. Selective scan: register-resident
// sequential kernel fused with softplus, u*D skip and silu(z) gating.
// ---------------------------------------------------------------------------

typedef __attribute__((ext_vector_type(16))) _Float16 v16h;
typedef __attribute__((ext_vector_type(8)))  _Float16 v8h;
typedef __attribute__((ext_vector_type(8)))  float    v8f;

#define EMB   128
#define NB    256     // batch
#define DI    256     // EXPAND*EMB
#define NST   16      // d_state
#define DTR   8       // dt_rank

#if defined(__has_builtin)
#if __has_builtin(__builtin_amdgcn_sched_barrier)
#define SCHED_FENCE() __builtin_amdgcn_sched_barrier(0)
#endif
#endif
#ifndef SCHED_FENCE
#define SCHED_FENCE() asm volatile("" ::: "memory")
#endif

static __device__ __forceinline__ float silu_f(float x) {
  return x / (1.f + __expf(-x));
}

// ------------------------- B packing ---------------------------------------
// Bt[n, k] = B[k, n] as f16; n in [0,Np), k in [0,Kp); zero padded.
__global__ void pack_bt_k(const float* __restrict__ B, _Float16* __restrict__ Bt,
                          int K, int N, int Kp, int Np) {
  long idx = (long)blockIdx.x * blockDim.x + threadIdx.x;
  if (idx >= (long)Np * Kp) return;
  int k = (int)(idx % Kp);
  int n = (int)(idx / Kp);
  Bt[idx] = (n < N && k < K) ? (_Float16)B[(size_t)k * N + n] : (_Float16)0.f;
}

// ------------------------- WMMA GEMM ---------------------------------------
// C[M,N] = A[M,K] @ B[K,N] (+bias). A fp32 row-major; B pre-packed as Bt.
// Block = 128 threads (4 waves); wave computes 16 rows x 64 cols.

static __device__ __forceinline__ void wmma_step(const v16h& a,
                                                 const _Float16* __restrict__ bp0,
                                                 long ntStride, long koff,
                                                 v8f& c0, v8f& c1, v8f& c2, v8f& c3) {
  // Issue all eight b128 B loads first; the sched barrier keeps them grouped
  // (fragments stay live in registers), then the four WMMAs run back-to-back
  // with staggered waits instead of four serialized load->wait->wmma chains.
  v16h bf0, bf1, bf2, bf3;
#pragma unroll
  for (int nt = 0; nt < 4; ++nt) {
    const _Float16* bp = bp0 + nt * ntStride + koff;
    const v8h lo = *(const v8h*)bp;
    const v8h hi = *(const v8h*)(bp + 16);
    v16h* dst = (nt == 0) ? &bf0 : (nt == 1) ? &bf1 : (nt == 2) ? &bf2 : &bf3;
#pragma unroll
    for (int i = 0; i < 8; ++i) { (*dst)[i] = lo[i]; (*dst)[8 + i] = hi[i]; }
  }
  SCHED_FENCE();
  c0 = __builtin_amdgcn_wmma_f32_16x16x32_f16(false, a, false, bf0, (short)0, c0, false, false);
  c1 = __builtin_amdgcn_wmma_f32_16x16x32_f16(false, a, false, bf1, (short)0, c1, false, false);
  c2 = __builtin_amdgcn_wmma_f32_16x16x32_f16(false, a, false, bf2, (short)0, c2, false, false);
  c3 = __builtin_amdgcn_wmma_f32_16x16x32_f16(false, a, false, bf3, (short)0, c3, false, false);
}

__global__ void gemm_f16_wmma(const float* __restrict__ A,
                              const _Float16* __restrict__ Bt,
                              const float* __restrict__ bias,
                              float* __restrict__ C,
                              int M, int N, int K, int Kp) {
  const int lane = threadIdx.x & 31;
  const int wave = threadIdx.x >> 5;
  const int m0 = (blockIdx.y * 4 + wave) * 16;
  const int n0 = blockIdx.x * 64;
  if (m0 >= M) return;                       // wave-uniform
  const int g  = lane >> 4;
  const int hl = lane & 15;
  int mrow = m0 + hl; if (mrow >= M) mrow = M - 1;   // address clamp only
  const float* __restrict__ Arow = A + (size_t)mrow * (size_t)K;
  const _Float16* __restrict__ bp0 = Bt + (size_t)(n0 + hl) * (size_t)Kp + g * 8;
  const long ntStride = (long)16 * Kp;

  v8f c0 = {0,0,0,0,0,0,0,0};
  v8f c1 = {0,0,0,0,0,0,0,0};
  v8f c2 = {0,0,0,0,0,0,0,0};
  v8f c3 = {0,0,0,0,0,0,0,0};

  const int Kmain = K & ~31;
  if ((K & 3) == 0) {
    // vectorized A path (rows 16B-aligned): 4x float4 per k-block
    for (int kb = 0; kb < Kmain; kb += 32) {
      const float4 a0 = *(const float4*)(Arow + kb + g * 8);
      const float4 a1 = *(const float4*)(Arow + kb + g * 8 + 4);
      const float4 a2 = *(const float4*)(Arow + kb + 16 + g * 8);
      const float4 a3 = *(const float4*)(Arow + kb + 16 + g * 8 + 4);
      v16h a;
      a[0]=(_Float16)a0.x;  a[1]=(_Float16)a0.y;  a[2]=(_Float16)a0.z;  a[3]=(_Float16)a0.w;
      a[4]=(_Float16)a1.x;  a[5]=(_Float16)a1.y;  a[6]=(_Float16)a1.z;  a[7]=(_Float16)a1.w;
      a[8]=(_Float16)a2.x;  a[9]=(_Float16)a2.y;  a[10]=(_Float16)a2.z; a[11]=(_Float16)a2.w;
      a[12]=(_Float16)a3.x; a[13]=(_Float16)a3.y; a[14]=(_Float16)a3.z; a[15]=(_Float16)a3.w;
      wmma_step(a, bp0, ntStride, kb, c0, c1, c2, c3);
    }
  } else {
    for (int kb = 0; kb < Kmain; kb += 32) {
      v16h a;
#pragma unroll
      for (int i = 0; i < 8; ++i) {
        a[i]     = (_Float16)Arow[kb + g * 8 + i];
        a[8 + i] = (_Float16)Arow[kb + 16 + g * 8 + i];
      }
      wmma_step(a, bp0, ntStride, kb, c0, c1, c2, c3);
    }
  }
  if (Kmain < K) {
    // K tail: clamp k (loads stay unconditional); Bt zero-padding zeroes the
    // contribution of any k >= K.
    v16h a;
#pragma unroll
    for (int i = 0; i < 8; ++i) {
      int k0 = Kmain + g * 8 + i;      if (k0 >= K) k0 = K - 1;
      int k1 = Kmain + 16 + g * 8 + i; if (k1 >= K) k1 = K - 1;
      a[i]     = (_Float16)Arow[k0];
      a[8 + i] = (_Float16)Arow[k1];
    }
    wmma_step(a, bp0, ntStride, Kmain, c0, c1, c2, c3);
  }

  // store: lane l -> col n0+nt*16+(l%16); vgpr r -> row m0 + r + (l/16)*8
#pragma unroll
  for (int nt = 0; nt < 4; ++nt) {
    const int ncol = n0 + nt * 16 + hl;
    if (ncol >= N) continue;
    const float bv = bias ? bias[ncol] : 0.f;
    const v8f ac = (nt == 0) ? c0 : (nt == 1) ? c1 : (nt == 2) ? c2 : c3;
#pragma unroll
    for (int r = 0; r < 8; ++r) {
      const int m = m0 + r + g * 8;
      if (m < M) C[(size_t)m * N + ncol] = ac[r] + bv;
    }
  }
}

// ------------------------- elementwise kernels -----------------------------

__global__ void inv_perm_k(const int* __restrict__ fwd, int* __restrict__ back,
                           int comp, int Bn) {
  long idx = (long)blockIdx.x * blockDim.x + threadIdx.x;
  if (idx >= (long)comp * Bn) return;
  int i = (int)(idx / Bn), b = (int)(idx % Bn);
  back[(size_t)fwd[idx] * Bn + b] = i;
}

__global__ void emb_gather_k(const float* __restrict__ ytok,
                             const int* __restrict__ fwd,
                             const float* __restrict__ embW,
                             const float* __restrict__ embb,
                             const float* __restrict__ pos,
                             const float* __restrict__ cls,
                             float* __restrict__ X,
                             int Bn, int L, int emb_in, int comp) {
  long idx = (long)blockIdx.x * blockDim.x + threadIdx.x;
  long total = (long)Bn * L * EMB;
  if (idx >= total) return;
  int c = (int)(idx & (EMB - 1));
  long r = idx >> 7;
  int l = (int)(r % L), b = (int)(r / L);
  float v;
  if (l == 0) {
    v = cls[c];
  } else {
    int j = fwd[(size_t)(l - 1) * Bn + b];
    const float* y = ytok + (size_t)b * ((size_t)comp * emb_in) + (size_t)j * emb_in;
    float s = embb[c];
    for (int e = 0; e < emb_in; ++e) s += y[e] * embW[e * EMB + c];
    v = s + pos[(size_t)j * EMB + c];
  }
  X[idx] = v;
}

__global__ void conv_silu_k(const float* __restrict__ xz,
                            const float* __restrict__ cw,
                            const float* __restrict__ cb,
                            float* __restrict__ U, int Bn, int L) {
  long idx = (long)blockIdx.x * blockDim.x + threadIdx.x;
  long total = (long)Bn * L * DI;
  if (idx >= total) return;
  int d = (int)(idx % DI);
  long r = idx / DI;
  int t = (int)(r % L), b = (int)(r / L);
  float s = cb[d];
#pragma unroll
  for (int j = 0; j < 4; ++j) {
    int tt = t - 3 + j;
    if (tt >= 0) s += cw[d * 4 + j] * xz[((size_t)b * L + tt) * (2 * DI) + d];
  }
  U[idx] = silu_f(s);
}

__global__ void __launch_bounds__(DI)
mamba_scan_k(const float* __restrict__ xdb,   // (B,L,40): dt|B|C
             const float* __restrict__ xz,    // (B,L,512): z at [DI..)
             const float* __restrict__ A_log,
             const float* __restrict__ Dp,
             const float* __restrict__ W_dt,
             const float* __restrict__ b_dt,
             float* __restrict__ U,           // in: u, out: y*silu(z)
             int L) {
  const int b = blockIdx.x;
  const int d = threadIdx.x;
  float Ar[NST];
#pragma unroll
  for (int n = 0; n < NST; ++n) Ar[n] = -__expf(A_log[d * NST + n]);
  float wdt[DTR];
#pragma unroll
  for (int r = 0; r < DTR; ++r) wdt[r] = W_dt[r * DI + d];
  const float Dd = Dp[d];
  const float bdt = b_dt[d];
  float h[NST];
#pragma unroll
  for (int n = 0; n < NST; ++n) h[n] = 0.f;

  for (int t = 0; t < L; ++t) {
    const size_t base = (size_t)b * L + t;
    const float* xr = xdb + base * 40;     // block-uniform -> scalar loads
    if (t + 1 < L) __builtin_prefetch(xdb + (base + 1) * 40, 0, 1);
    float dtv = bdt;
#pragma unroll
    for (int r = 0; r < DTR; ++r) dtv += xr[r] * wdt[r];
    const float delta = (dtv > 20.f) ? dtv : log1pf(__expf(dtv));
    const float u = U[base * DI + d];
    const float z = xz[base * (2 * DI) + DI + d];
    const float du = delta * u;
    float acc = 0.f;
#pragma unroll
    for (int n = 0; n < NST; ++n) {
      h[n] = __expf(delta * Ar[n]) * h[n] + du * xr[8 + n];
      acc += h[n] * xr[24 + n];
    }
    float y = acc + u * Dd;
    y *= silu_f(z);
    U[base * DI + d] = y;
  }
}

__global__ void layernorm_k(float* __restrict__ X, const float* __restrict__ g,
                            const float* __restrict__ bb, long rows) {
  const int lane = threadIdx.x & 31;
  const long row = (long)blockIdx.x * 4 + (threadIdx.x >> 5);
  if (row >= rows) return;
  float4 v = ((const float4*)(X + row * EMB))[lane];
  float s = v.x + v.y + v.z + v.w;
#pragma unroll
  for (int o = 16; o > 0; o >>= 1) s += __shfl_xor(s, o, 32);
  const float mu = s * (1.f / EMB);
  const float dx = v.x - mu, dy = v.y - mu, dz = v.z - mu, dw = v.w - mu;
  float q = dx * dx + dy * dy + dz * dz + dw * dw;
#pragma unroll
  for (int o = 16; o > 0; o >>= 1) q += __shfl_xor(q, o, 32);
  const float inv = rsqrtf(q * (1.f / EMB) + 1e-5f);
  const float4 gg = ((const float4*)g)[lane];
  const float4 b4 = ((const float4*)bb)[lane];
  float4 o4;
  o4.x = dx * inv * gg.x + b4.x;
  o4.y = dy * inv * gg.y + b4.y;
  o4.z = dz * inv * gg.z + b4.z;
  o4.w = dw * inv * gg.w + b4.w;
  ((float4*)(X + row * EMB))[lane] = o4;
}

__global__ void dec_gather_k(const float* __restrict__ feat,
                             const int* __restrict__ back,
                             const float* __restrict__ maskTok,
                             const float* __restrict__ pos,
                             float* __restrict__ X,
                             int Bn, int T, int Tvis) {
  const int L = T + 1;
  long idx = (long)blockIdx.x * blockDim.x + threadIdx.x;
  long total = (long)Bn * L * EMB;
  if (idx >= total) return;
  int c = (int)(idx & (EMB - 1));
  long r = idx >> 7;
  int t = (int)(r % L), b = (int)(r / L);
  int bi = (t == 0) ? 0 : (back[(size_t)(t - 1) * Bn + b] + 1);
  float v = (bi < Tvis) ? feat[((size_t)b * Tvis + bi) * EMB + c] : maskTok[c];
  X[idx] = v + pos[(size_t)t * EMB + c];
}

__global__ void dec_cls_k(const float* __restrict__ f, float* __restrict__ rc,
                          int Bn, int Lr) {
  long idx = (long)blockIdx.x * blockDim.x + threadIdx.x;
  if (idx >= (long)Bn * EMB) return;
  int b = (int)(idx >> 7), c = (int)(idx & (EMB - 1));
  rc[idx] = f[(size_t)b * Lr * EMB + c];
}

__global__ void dec_pat_k(const float* __restrict__ f,
                          const float* __restrict__ decW,
                          const float* __restrict__ decb,
                          const int* __restrict__ back,
                          float* __restrict__ rp, float* __restrict__ rm,
                          int Bn, int T, int e, int Tvis) {
  long idx = (long)blockIdx.x * blockDim.x + threadIdx.x;
  if (idx >= (long)Bn * T * e) return;
  int j = (int)(idx % e);
  int t = (int)((idx / e) % T);
  int b = (int)(idx / ((long)e * T));
  const float* row = f + ((size_t)b * (T + 1) + (t + 1)) * EMB;
  float s = decb[j];
#pragma unroll 4
  for (int c = 0; c < EMB; ++c) s += row[c] * decW[c * e + j];
  rp[idx] = s;
  rm[idx] = (back[(size_t)t * Bn + b] >= (Tvis - 1)) ? 1.f : 0.f;
}

// ------------------------- host orchestration ------------------------------

struct MambaP {
  const float *A_log, *D, *W_dt, *W_in, *W_out, *W_x, *b_dt, *conv_b, *conv_w;
};

static inline unsigned cdivu(long a, long b) { return (unsigned)((a + b - 1) / b); }

static void gemm(const float* A, const float* B, const float* bias, float* C,
                 long M, int N, int K, _Float16* Bt, hipStream_t s) {
  const int Kp = (K + 31) & ~31;
  const int Np = (int)cdivu(N, 64) * 64;
  pack_bt_k<<<cdivu((long)Np * Kp, 256), 256, 0, s>>>(B, Bt, K, N, Kp, Np);
  dim3 g(Np / 64, cdivu(M, 64));
  gemm_f16_wmma<<<g, 128, 0, s>>>(A, Bt, bias, C, (int)M, N, K, Kp);
}

static void launch_mamba(const float* X, float* XZ, float* U, float* XDB,
                         float* O, const MambaP& mp, int L, _Float16* Bt,
                         hipStream_t s) {
  const long M = (long)NB * L;
  gemm(X, mp.W_in, nullptr, XZ, M, 2 * DI, EMB, Bt, s);                // in-proj
  conv_silu_k<<<cdivu(M * DI, 256), 256, 0, s>>>(XZ, mp.conv_w, mp.conv_b, U, NB, L);
  gemm(U, mp.W_x, nullptr, XDB, M, DTR + 2 * NST, DI, Bt, s);          // x-proj
  mamba_scan_k<<<NB, DI, 0, s>>>(XDB, XZ, mp.A_log, mp.D, mp.W_dt, mp.b_dt, U, L);
  gemm(U, mp.W_out, nullptr, O, M, EMB, DI, Bt, s);                    // out-proj
}

static MambaP mk_mp(const float* const* lv) {
  MambaP m;
  m.A_log = lv[0]; m.D = lv[1]; m.W_dt = lv[2]; m.W_in = lv[3]; m.W_out = lv[4];
  m.W_x = lv[5]; m.b_dt = lv[6]; m.conv_b = lv[7]; m.conv_w = lv[8];
  return m;
}

// encoder leaves (sorted keys): cls, emb_W, emb_b, ln_b, ln_g,
//   [A_log, D, W_dt, W_in, W_out, W_x, b_dt, conv_b, conv_w], pos, tok_W, tok_b
// decoder leaves: dec_W, dec_b, [mamba...], mask, pos
static void run_modality(const float* patches, const int* fwd,
                         const float* const* encL, const float* const* decL,
                         int comp, int emb_in,
                         float* outP, float* outM, float* outC,
                         float* ytok, float* X, float* XZ, float* U,
                         float* XDB, float* O, int* back, _Float16* Bt,
                         hipStream_t s) {
  const int tokens = comp * emb_in;
  const int remain = (int)(comp * (1.0 - 0.15));
  const int L1 = remain + 1;
  const int L2 = comp + 1;
  const MambaP em = mk_mp(encL + 5);
  const MambaP dm = mk_mp(decL + 2);

  gemm(patches, encL[15], encL[16], ytok, NB, tokens, tokens, Bt, s);  // tokenizer
  inv_perm_k<<<cdivu((long)comp * NB, 256), 256, 0, s>>>(fwd, back, comp, NB);
  emb_gather_k<<<cdivu((long)NB * L1 * EMB, 256), 256, 0, s>>>(
      ytok, fwd, encL[1], encL[2], encL[14], encL[0], X, NB, L1, emb_in, comp);
  launch_mamba(X, XZ, U, XDB, O, em, L1, Bt, s);
  layernorm_k<<<cdivu((long)NB * L1, 4), 128, 0, s>>>(O, encL[4], encL[3],
                                                      (long)NB * L1);
  dec_gather_k<<<cdivu((long)NB * L2 * EMB, 256), 256, 0, s>>>(
      O, back, decL[11], decL[12], X, NB, comp, L1);
  launch_mamba(X, XZ, U, XDB, O, dm, L2, Bt, s);
  dec_cls_k<<<cdivu((long)NB * EMB, 256), 256, 0, s>>>(O, outC, NB, L2);
  dec_pat_k<<<cdivu((long)NB * comp * emb_in, 256), 256, 0, s>>>(
      O, decL[0], decL[1], back, outP, outM, NB, comp, emb_in, L1);
}

static void fill_blob(const float* base, const long* sz, int n, const float** out) {
  long off = 0;
  for (int i = 0; i < n; ++i) { out[i] = base + off; off += sz[i]; }
}

extern "C" void kernel_launch(void* const* d_in, const int* in_sizes, int n_in,
                              void* d_out, int out_size, void* d_ws, size_t ws_size,
                              hipStream_t stream) {
  (void)in_sizes; (void)out_size; (void)ws_size;
  const float* patches1 = (const float*)d_in[0];
  const float* patches2 = (const float*)d_in[1];
  const int* fwd1 = (const int*)d_in[2];
  const int* fwd2 = (const int*)d_in[3];

  static const long rna_enc_sz[17] = {128, 256, 128, 128, 128,
                                      4096, 256, 2048, 65536, 32768, 10240, 256, 256, 1024,
                                      256000, 16000000, 4000};
  static const long adt_enc_sz[17] = {128, 128, 128, 128, 128,
                                      4096, 256, 2048, 65536, 32768, 10240, 256, 256, 1024,
                                      1792, 196, 14};
  static const long rna_dec_sz[13] = {256, 2,
                                      4096, 256, 2048, 65536, 32768, 10240, 256, 256, 1024,
                                      128, 256128};
  static const long adt_dec_sz[13] = {128, 1,
                                      4096, 256, 2048, 65536, 32768, 10240, 256, 256, 1024,
                                      128, 1920};

  const float* encR[17]; const float* encA[17];
  const float* decR[13]; const float* decA[13];
  if (n_in >= 58) {  // one input per pytree leaf
    int k = 4;
    for (int i = 0; i < 17; ++i) encR[i] = (const float*)d_in[k++];
    for (int i = 0; i < 17; ++i) encA[i] = (const float*)d_in[k++];
    for (int i = 0; i < 13; ++i) decR[i] = (const float*)d_in[k++];
    for (int i = 0; i < 13; ++i) decA[i] = (const float*)d_in[k++];
  } else {           // one flattened blob per top-level dict entry
    fill_blob((const float*)d_in[4], rna_enc_sz, 17, encR);
    fill_blob((const float*)d_in[5], adt_enc_sz, 17, encA);
    fill_blob((const float*)d_in[6], rna_dec_sz, 13, decR);
    fill_blob((const float*)d_in[7], adt_dec_sz, 13, decA);
  }

  // workspace carve (floats), sized for max L = 2001
  const long Lmax = 2001;
  float* ws = (float*)d_ws;
  float* ytok = ws;                         long o = (long)NB * 4000;
  float* X    = ws + o;                     o += (long)NB * Lmax * EMB;
  float* XZ   = ws + o;                     o += (long)NB * Lmax * 2 * DI;
  float* U    = ws + o;                     o += (long)NB * Lmax * DI;
  float* XDB  = ws + o;                     o += (long)NB * Lmax * 40;
  float* O    = ws + o;                     o += (long)NB * Lmax * EMB;
  int* back   = (int*)(ws + o);             o += (long)2000 * NB;  // int == float size
  _Float16* Bt = (_Float16*)(ws + o);       // max Np*Kp = 4032*4000 halves (~32MB)

  float* out = (float*)d_out;
  float* rp = out;                       // 256*4000
  float* rm = rp + (long)NB * 4000;      // 256*4000
  float* rc = rm + (long)NB * 4000;      // 256*128
  float* ap = rc + (long)NB * EMB;       // 256*14
  float* am = ap + (long)NB * 14;        // 256*14
  float* ac = am + (long)NB * 14;        // 256*128

  run_modality(patches1, fwd1, encR, decR, 2000, 2, rp, rm, rc,
               ytok, X, XZ, U, XDB, O, back, Bt, stream);
  run_modality(patches2, fwd2, encA, decA, 14, 1, ap, am, ac,
               ytok, X, XZ, U, XDB, O, back, Bt, stream);
}
